// GNNProfileDetector_14474039788040
// MI455X (gfx1250) — matrix operand order repaired and checked by reading, not compiled, not run on hardware
//
#include <hip/hip_runtime.h>
#include <stdint.h>
#include <math.h>

// Problem constants (match reference)
#define NODES   100000
#define EDGES   1600000
#define F_INF   256
#define HDIM    128
#define NGRAPH  1000

typedef __attribute__((ext_vector_type(16))) __bf16 v16bf;
typedef __attribute__((ext_vector_type(8)))  float  v8f;

static __device__ __forceinline__ unsigned short f32_to_bf16_rne(float f) {
    unsigned int u = __float_as_uint(f);
    u += 0x7FFFu + ((u >> 16) & 1u);          // round-to-nearest-even
    return (unsigned short)(u >> 16);
}

// ---------------------------------------------------------------------------
// Degree / normalization
// ---------------------------------------------------------------------------
__global__ void k_deg_init(float* deg, int n) {
    int i = blockIdx.x * blockDim.x + threadIdx.x;
    if (i < n) deg[i] = 1.0f;                  // self-loop contribution
}

__global__ void k_deg_edges(const int* __restrict__ dst, float* deg, int e) {
    int i = blockIdx.x * blockDim.x + threadIdx.x;
    if (i < e) atomicAdd(&deg[dst[i]], 1.0f);
}

__global__ void k_dis(const float* __restrict__ deg, float* dis, int n) {
    int i = blockIdx.x * blockDim.x + threadIdx.x;
    if (i < n) {
        float d = deg[i];
        dis[i] = (d > 0.0f) ? rsqrtf(d) : 0.0f;
    }
}

// ---------------------------------------------------------------------------
// Elementwise f32 -> bf16 (row-major, same layout)
// ---------------------------------------------------------------------------
__global__ void k_cvt_bf16(const float* __restrict__ in, unsigned short* __restrict__ out,
                           int n) {
    int i = blockIdx.x * blockDim.x + threadIdx.x;
    if (i < n) out[i] = f32_to_bf16_rne(in[i]);
}

// ---------------------------------------------------------------------------
// Weight convert + transpose: W[K x H] f32  ->  Wt[H x K] bf16
// ---------------------------------------------------------------------------
__global__ void k_cvt_w_t(const float* __restrict__ W, unsigned short* __restrict__ Wt,
                          int K, int H) {
    int idx = blockIdx.x * blockDim.x + threadIdx.x;
    if (idx < K * H) {
        int k = idx / H;
        int n = idx - k * H;
        Wt[(size_t)n * K + k] = f32_to_bf16_rne(W[idx]);
    }
}

// ---------------------------------------------------------------------------
// GEMM: C[N x 128] = A[N x K](bf16 row-major) @ Bt[128 x K](bf16, W transposed)
// 256 threads = 8 waves per block. The whole weight matrix Bt is staged in
// LDS once per block (64 KB @ K=256, 32 KB @ K=128; WGP has 320 KB), then
// each wave computes a 16 x 128 row-strip: the A fragment (global bf16,
// b128 loads) is reused by 8 column-tile WMMAs whose B fragments come from
// LDS (ds_load_b128). K compile-time -> fully unrolled v_wmma stream.
// EXEC is all-ones for every wave reaching the WMMAs (wave-uniform guard).
// ---------------------------------------------------------------------------
template <int K>
__global__ __launch_bounds__(256)
void k_gemm_wmma_bf16(const unsigned short* __restrict__ A,
                      const unsigned short* __restrict__ Bt,
                      float* __restrict__ C, int n_tiles) {
    __shared__ unsigned short bs[HDIM * K];   // transposed weights, bf16

    // Cooperative staging: 128*K bf16 = 16*K DWORD4s, 256 threads, b128 chunks
    const int tid = threadIdx.x;
    constexpr int total = HDIM * K;           // elements
    for (int i = tid * 8; i < total; i += 256 * 8)
        *(uint4*)(bs + i) = *(const uint4*)(Bt + i);
    __syncthreads();

    const int wave = tid >> 5;                // 0..7
    const int lane = tid & 31;
    const int tm   = blockIdx.x * 8 + wave;   // row tile (16 rows)
    if (tm >= n_tiles) return;                // wave-uniform; after the barrier

    const int half = lane >> 4;               // 0 / 1
    const int l15  = lane & 15;

    const unsigned short* arow = A + (size_t)(tm * 16 + l15) * K;

    v8f acc[8] = {};                          // 8 column tiles of 16 (H = 128)

#pragma unroll
    for (int k0 = 0; k0 < K; k0 += 32) {
        // A fragment (16-bit A 16x32 layout): lanes 0-15 K in {0..7,16..23},
        // lanes 16-31 shifted by +8; 2 bf16 per VGPR.
        union { unsigned int u[8]; v16bf v; } a;
#pragma unroll
        for (int v = 0; v < 8; ++v) {
            int ak = k0 + ((v < 4) ? (v * 2) : (16 + (v - 4) * 2)) + (half ? 8 : 0);
            a.u[v] = *(const unsigned int*)(arow + ak);
        }
#pragma unroll
        for (int t = 0; t < 8; ++t) {
            // B fragment (32x16): lane = column, lanes 0-15 hold K {0..7,16..23},
            // lanes 16-31 shifted by +8 (same K order as A).
            const unsigned short* bcol = bs + (t * 16 + l15) * K + k0 + half * 8;
            union { unsigned int u[8]; v16bf v; } b;
#pragma unroll
            for (int v = 0; v < 8; ++v)
                b.u[v] = *(const unsigned int*)(bcol + ((v < 4) ? (v * 2) : (16 + (v - 4) * 2)));
            acc[t] = __builtin_amdgcn_wmma_f32_16x16x32_bf16(
                         false, a.v, false, b.v, (short)0, acc[t], false, false);
        }
    }

    // C/D layout: VGPR r -> M=r (lanes 0-15) / M=r+8 (lanes 16-31), N=l15
    float* cbase = C + (size_t)(tm * 16) * HDIM;
#pragma unroll
    for (int t = 0; t < 8; ++t) {
#pragma unroll
        for (int r = 0; r < 8; ++r) {
            int m = r + (half ? 8 : 0);
            cbase[(size_t)m * HDIM + t * 16 + l15] = acc[t][r];
        }
    }
}

// ---------------------------------------------------------------------------
// acc[i,:] = lin[i,:] * dis[i]^2   (self-loop message; also initializes acc)
// ---------------------------------------------------------------------------
__global__ void k_init_acc(const float* __restrict__ lin, const float* __restrict__ dis,
                           float* __restrict__ acc, int n) {
    int idx = blockIdx.x * blockDim.x + threadIdx.x;   // over NODES * HDIM
    if (idx < n) {
        int i = idx >> 7;                              // / HDIM
        float s = dis[i];
        acc[idx] = lin[idx] * s * s;
    }
}

// ---------------------------------------------------------------------------
// Edge scatter: acc[dst] += lin[src] * (dis[src]*dis[dst]), float4 chunks
// ---------------------------------------------------------------------------
__global__ void k_scatter(const int* __restrict__ src, const int* __restrict__ dst,
                          const float* __restrict__ dis,
                          const float* __restrict__ lin, float* __restrict__ acc,
                          int e) {
    int idx = blockIdx.x * blockDim.x + threadIdx.x;   // over EDGES * (HDIM/4)
    int ei = idx >> 5;                                 // 32 chunks of 4 per edge
    if (ei >= e) return;
    int c  = (idx & 31) << 2;
    int s  = src[ei];
    int d  = dst[ei];
    float nrm = dis[s] * dis[d];
    const float4 v = *(const float4*)(lin + (size_t)s * HDIM + c);
    float* p = acc + (size_t)d * HDIM + c;
    atomicAdd(p + 0, v.x * nrm);
    atomicAdd(p + 1, v.y * nrm);
    atomicAdd(p + 2, v.z * nrm);
    atomicAdd(p + 3, v.w * nrm);
}

// ---------------------------------------------------------------------------
// bias + ReLU -> bf16 (feeds next layer's GEMM directly)
// ---------------------------------------------------------------------------
__global__ void k_bias_relu_bf16(const float* __restrict__ h, const float* __restrict__ b,
                                 unsigned short* __restrict__ out, int n) {
    int idx = blockIdx.x * blockDim.x + threadIdx.x;   // over NODES * HDIM
    if (idx < n) {
        int j = idx & (HDIM - 1);
        out[idx] = f32_to_bf16_rne(fmaxf(h[idx] + b[j], 0.0f));
    }
}

// ---------------------------------------------------------------------------
// In-place bias + ReLU (f32, last layer -> pooling)
// ---------------------------------------------------------------------------
__global__ void k_bias_relu(float* __restrict__ h, const float* __restrict__ b, int n) {
    int idx = blockIdx.x * blockDim.x + threadIdx.x;   // over NODES * HDIM
    if (idx < n) {
        int j = idx & (HDIM - 1);
        h[idx] = fmaxf(h[idx] + b[j], 0.0f);
    }
}

// ---------------------------------------------------------------------------
// Pooling
// ---------------------------------------------------------------------------
__global__ void k_zero(float* p, int n) {
    int i = blockIdx.x * blockDim.x + threadIdx.x;
    if (i < n) p[i] = 0.0f;
}

__global__ void k_pool_feat(const float* __restrict__ h, const int* __restrict__ batch,
                            float* __restrict__ sums, int n) {
    int idx = blockIdx.x * blockDim.x + threadIdx.x;   // over NODES * (HDIM/4)
    int i = idx >> 5;
    if (i >= n) return;
    int c = (idx & 31) << 2;
    int g = batch[i];
    const float4 v = *(const float4*)(h + (size_t)i * HDIM + c);
    float* p = sums + (size_t)g * HDIM + c;
    atomicAdd(p + 0, v.x);
    atomicAdd(p + 1, v.y);
    atomicAdd(p + 2, v.z);
    atomicAdd(p + 3, v.w);
}

__global__ void k_pool_cnt(const int* __restrict__ batch, float* __restrict__ cnt, int n) {
    int i = blockIdx.x * blockDim.x + threadIdx.x;
    if (i < n) atomicAdd(&cnt[batch[i]], 1.0f);
}

// ---------------------------------------------------------------------------
// Head: mean -> fc1(128->32) relu -> fc2(32->2) -> log_softmax
// One 32-thread block per graph.
// ---------------------------------------------------------------------------
__global__ __launch_bounds__(32)
void k_head(const float* __restrict__ sums, const float* __restrict__ cnt,
            const float* __restrict__ fw1, const float* __restrict__ fb1,
            const float* __restrict__ fw2, const float* __restrict__ fb2,
            float* __restrict__ out) {
    int g    = blockIdx.x;
    int lane = threadIdx.x;                 // 0..31 -> hidden unit
    __shared__ float hid[32];

    float c   = fmaxf(cnt[g], 1.0f);
    float inv = 1.0f / c;
    const float* srow = sums + (size_t)g * HDIM;
    float a = fb1[lane];
    for (int k = 0; k < HDIM; ++k)
        a += (srow[k] * inv) * fw1[k * 32 + lane];
    hid[lane] = fmaxf(a, 0.0f);
    __syncthreads();

    if (lane < 2) {
        float z = fb2[lane];
        for (int k = 0; k < 32; ++k) z += hid[k] * fw2[k * 2 + lane];
        hid[lane] = z;                      // reuse LDS for both logits
    }
    __syncthreads();
    if (lane < 2) {
        float z0 = hid[0], z1 = hid[1];
        float m  = fmaxf(z0, z1);
        float lse = m + logf(expf(z0 - m) + expf(z1 - m));
        out[g * 2 + lane] = hid[lane] - lse;
    }
}

// ---------------------------------------------------------------------------
// Launch
// ---------------------------------------------------------------------------
extern "C" void kernel_launch(void* const* d_in, const int* in_sizes, int n_in,
                              void* d_out, int out_size, void* d_ws, size_t ws_size,
                              hipStream_t stream) {
    const float* x     = (const float*)d_in[0];
    const int*   ei    = (const int*)  d_in[1];
    const int*   batch = (const int*)  d_in[2];
    const float* W1 = (const float*)d_in[3];
    const float* b1 = (const float*)d_in[4];
    const float* W2 = (const float*)d_in[5];
    const float* b2 = (const float*)d_in[6];
    const float* W3 = (const float*)d_in[7];
    const float* b3 = (const float*)d_in[8];
    const float* fw1 = (const float*)d_in[9];
    const float* fb1 = (const float*)d_in[10];
    const float* fw2 = (const float*)d_in[11];
    const float* fb2 = (const float*)d_in[12];
    const int* src = ei;            // edge_index[0]
    const int* dst = ei + EDGES;    // edge_index[1]

    // Workspace carve-out (256B aligned)
    char* base = (char*)d_ws;
    size_t off = 0;
    auto carve = [&](size_t bytes) -> char* {
        char* p = base + off;
        off = (off + bytes + 255) & ~(size_t)255;
        return p;
    };
    float*          deg  = (float*)carve((size_t)NODES * 4);
    float*          dis  = (float*)carve((size_t)NODES * 4);
    float*          bufA = (float*)carve((size_t)NODES * HDIM * 4);   // GEMM out (lin)
    float*          bufB = (float*)carve((size_t)NODES * HDIM * 4);   // acc / h3
    unsigned short* xbf  = (unsigned short*)carve((size_t)NODES * F_INF * 2);
    unsigned short* hbf  = (unsigned short*)carve((size_t)NODES * HDIM * 2);
    unsigned short* w1t  = (unsigned short*)carve((size_t)F_INF * HDIM * 2);
    unsigned short* w2t  = (unsigned short*)carve((size_t)HDIM  * HDIM * 2);
    unsigned short* w3t  = (unsigned short*)carve((size_t)HDIM  * HDIM * 2);
    float*          sums = (float*)carve((size_t)NGRAPH * HDIM * 4);
    float*          cnt  = (float*)carve((size_t)NGRAPH * 4);
    (void)ws_size; (void)in_sizes; (void)n_in; (void)out_size;

    const int NH  = NODES * HDIM;           // 12.8M
    const int B   = 256;
    auto gb = [](int n, int b) { return (n + b - 1) / b; };

    // Weights -> bf16 transposed; x -> bf16
    k_cvt_w_t<<<gb(F_INF * HDIM, B), B, 0, stream>>>(W1, w1t, F_INF, HDIM);
    k_cvt_w_t<<<gb(HDIM  * HDIM, B), B, 0, stream>>>(W2, w2t, HDIM, HDIM);
    k_cvt_w_t<<<gb(HDIM  * HDIM, B), B, 0, stream>>>(W3, w3t, HDIM, HDIM);
    k_cvt_bf16<<<gb(NODES * F_INF, B), B, 0, stream>>>(x, xbf, NODES * F_INF);

    // Degree / norm
    k_deg_init <<<gb(NODES, B), B, 0, stream>>>(deg, NODES);
    k_deg_edges<<<gb(EDGES, B), B, 0, stream>>>(dst, deg, EDGES);
    k_dis      <<<gb(NODES, B), B, 0, stream>>>(deg, dis, NODES);

    const int n_tiles     = NODES / 16;              // 6250 row strips
    const int gemm_blocks = gb(n_tiles, 8);          // 8 waves per block
    const int scat_n      = EDGES * (HDIM / 4);      // 51.2M threads

    // ---- Layer 1: K = 256 ----
    k_gemm_wmma_bf16<F_INF><<<gemm_blocks, 256, 0, stream>>>(xbf, w1t, bufA, n_tiles);
    k_init_acc <<<gb(NH, B), B, 0, stream>>>(bufA, dis, bufB, NH);
    k_scatter  <<<gb(scat_n, B), B, 0, stream>>>(src, dst, dis, bufA, bufB, EDGES);
    k_bias_relu_bf16<<<gb(NH, B), B, 0, stream>>>(bufB, b1, hbf, NH);

    // ---- Layer 2: K = 128 ----
    k_gemm_wmma_bf16<HDIM><<<gemm_blocks, 256, 0, stream>>>(hbf, w2t, bufA, n_tiles);
    k_init_acc <<<gb(NH, B), B, 0, stream>>>(bufA, dis, bufB, NH);
    k_scatter  <<<gb(scat_n, B), B, 0, stream>>>(src, dst, dis, bufA, bufB, EDGES);
    k_bias_relu_bf16<<<gb(NH, B), B, 0, stream>>>(bufB, b2, hbf, NH);

    // ---- Layer 3: K = 128 ----
    k_gemm_wmma_bf16<HDIM><<<gemm_blocks, 256, 0, stream>>>(hbf, w3t, bufA, n_tiles);
    k_init_acc <<<gb(NH, B), B, 0, stream>>>(bufA, dis, bufB, NH);
    k_scatter  <<<gb(scat_n, B), B, 0, stream>>>(src, dst, dis, bufA, bufB, EDGES);
    k_bias_relu<<<gb(NH, B), B, 0, stream>>>(bufB, b3, NH);

    // ---- Pool + head ----
    k_zero<<<gb(NGRAPH * HDIM, B), B, 0, stream>>>(sums, NGRAPH * HDIM);
    k_zero<<<gb(NGRAPH, B), B, 0, stream>>>(cnt, NGRAPH);
    k_pool_feat<<<gb(NODES * (HDIM / 4), B), B, 0, stream>>>(bufB, batch, sums, NODES);
    k_pool_cnt <<<gb(NODES, B), B, 0, stream>>>(batch, cnt, NODES);
    k_head<<<NGRAPH, 32, 0, stream>>>(sums, cnt, fw1, fb1, fw2, fb2, (float*)d_out);
}
